// LongformerSelfAttentionWrapper_77936476553419
// MI455X (gfx1250) — compile-verified, matrix operand order checked by reading, not run on hardware
//
#include <hip/hip_runtime.h>

#if __has_include(<hip/amd_detail/amd_gfx1250_TDM.h>)
#define TDM_6ARG 1
#endif
#if __has_builtin(__builtin_amdgcn_tensor_load_to_lds) && __has_builtin(__builtin_amdgcn_s_wait_tensorcnt)
#define USE_TDM 1
#endif

// ---------- CDNA5 WMMA / TDM types ----------
typedef __attribute__((ext_vector_type(16))) __bf16 v16bf;
typedef __attribute__((ext_vector_type(8)))  float  v8f;
typedef __attribute__((ext_vector_type(4)))  unsigned int v4u;
typedef __attribute__((ext_vector_type(8)))  int v8i;
typedef __attribute__((ext_vector_type(4)))  int v4i;

union Frag {
    v16bf v;
    unsigned int u[8];
    unsigned short us[16];
};

// Constants for this problem
#define BB 2
#define TT 4096
#define EE 768
#define HH 12
#define DH 64
#define WW 256
#define NC 16
#define GG 16

// ---------- helpers ----------
__device__ __forceinline__ unsigned short f2bf(float f) {
    unsigned int u = __float_as_uint(f);
    unsigned int r = u + 0x7FFFu + ((u >> 16) & 1u);
    return (unsigned short)(r >> 16);
}

// Starting (even) K index of 16-bit element-pair p (p=0..7) for a lane-half L,
// matching the CDNA5 16-bit A/B fragment layout (16xK, K in {0..7,16..23}+8L).
__device__ __forceinline__ int kpair2(int p, int L) {
    return ((p & 4) << 2) + (L << 3) + ((p & 3) << 1);
}
// K index of single 16-bit element e (e=0..15) for lane-half L.
__device__ __forceinline__ int kloc(int e, int L) {
    return ((e & 8) << 1) + (L << 3) + (e & 7);
}

// reductions across the 16 lanes of each wave-half (halves stay independent)
__device__ __forceinline__ float rmax16(float x) {
#pragma unroll
    for (int m = 1; m < 16; m <<= 1) x = fmaxf(x, __shfl_xor(x, m, 32));
    return x;
}
__device__ __forceinline__ float rsum16(float x) {
#pragma unroll
    for (int m = 1; m < 16; m <<= 1) x += __shfl_xor(x, m, 32);
    return x;
}

#ifdef USE_TDM
// Issue a TDM 2D tile DMA: 64 rows x 32 cols of bf16 from a row-major
// 768-stride matrix (global, tile start = gaddr) into LDS at lds_off.
// D# per CDNA5 ISA ch.8: group0 {count=1, lds_addr, global_addr, type=2},
// group1 {data_size=2B, tensor_dim0=768, tensor_dim1=768, tile=32x64, stride0=768}.
__device__ __forceinline__ void tdm_load_tile(unsigned lds_off, unsigned long long gaddr) {
    v4u g0 = { 1u,
               lds_off,
               (unsigned)gaddr,
               (unsigned)((gaddr >> 32) & 0x01ffffffu) | (2u << 30) };
    v8i g1 = { 0x10000,                 // workgroup_mask=0, data_size=1 (2 bytes)
               (int)(768u << 16),       // tensor_dim0 lo16 in [31:16]
               (int)(768u << 16),       // tensor_dim0 hi16=0, tensor_dim1 lo16 in [31:16]
               (int)(32u << 16),        // tensor_dim1 hi16=0, tile_dim0=32 in [31:16]
               64,                      // tile_dim1=64, tile_dim2=0
               768,                     // tensor_dim0_stride lo32
               0, 0 };
    v4i gz = { 0, 0, 0, 0 };
#ifdef TDM_6ARG
    v8i gz8 = { 0, 0, 0, 0, 0, 0, 0, 0 };
    __builtin_amdgcn_tensor_load_to_lds(g0, g1, gz, gz, gz8, 0);
#else
    __builtin_amdgcn_tensor_load_to_lds(g0, g1, gz, gz, 0);
#endif
}
#endif

// ---------- conversion kernels ----------
__global__ void __launch_bounds__(256) conv_x_kernel(const float* __restrict__ x,
                                                     unsigned short* __restrict__ xb, int n) {
    int i = blockIdx.x * 256 + threadIdx.x;
    if (i < n) xb[i] = f2bf(x[i]);
}

__global__ void __launch_bounds__(256) conv_w_kernel(const float* W0, const float* W1,
                                                     const float* W2, const float* W3,
                                                     const float* W4, const float* W5,
                                                     unsigned short* __restrict__ wt) {
    int i = blockIdx.x * 256 + threadIdx.x;        // 0 .. 768*768-1
    int z = blockIdx.y;
    const float* W = (z == 0) ? W0 : (z == 1) ? W1 : (z == 2) ? W2
                   : (z == 3) ? W3 : (z == 4) ? W4 : W5;
    int nrow = i / EE, kcol = i - nrow * EE;
    // Wt[n][k] = W[k][n]  (bf16)
    wt[(size_t)z * EE * EE + i] = f2bf(W[kcol * EE + nrow]);
}

// ---------- fused projection GEMM: Y = (xb @ W + b) * scale, bf16 out ----------
// Block tile 256(M) x 64(N); 8 waves, each wave 32(M) x 64(N) = 8 WMMAs/K-step.
// B tile (Wt 64x32 bf16) double-buffered in LDS via Tensor Data Mover (TDM),
// issued by wave 0 and overlapped with WMMA compute; fallback: cooperative loads.
__global__ void __launch_bounds__(256)
proj_gemm_kernel(const unsigned short* __restrict__ xb, const unsigned short* __restrict__ wt,
                 unsigned short* __restrict__ proj,
                 const float* bq, const float* bk, const float* bv,
                 const float* bqg, const float* bkg, const float* bvg) {
    __shared__ __align__(16) unsigned short Bs[2][64 * 32];   // 2 x 4KB

    const int tid = threadIdx.x;
    const int wave = tid >> 5, lane = tid & 31, L = lane >> 4, hl = lane & 15;
    const int z = blockIdx.z;
    const unsigned short* w = wt + (size_t)z * EE * EE;
    unsigned short* out = proj + (size_t)z * (BB * TT) * EE;
    const float* bias = (z == 0) ? bq : (z == 1) ? bk : (z == 2) ? bv
                      : (z == 3) ? bqg : (z == 4) ? bkg : bvg;
    const float scalef = (z == 0 || z == 3) ? 0.125f : 1.0f;   // 1/sqrt(64) for q, qg

    const int n0 = blockIdx.x * 64;
    const int m0 = blockIdx.y * 256 + wave * 32;

    const unsigned int* adw0 = (const unsigned int*)xb + (((m0 + hl) * EE) >> 1);
    const unsigned int* adw1 = (const unsigned int*)xb + (((m0 + 16 + hl) * EE) >> 1);

    v8f acc[2][4] = {};

    auto compute_step = [&](int k0, const unsigned int* bs_dw) {
        Frag a0, a1;
#pragma unroll
        for (int p = 0; p < 8; ++p) {
            a0.u[p] = adw0[(k0 + kpair2(p, L)) >> 1];
            a1.u[p] = adw1[(k0 + kpair2(p, L)) >> 1];
        }
#pragma unroll
        for (int nt = 0; nt < 4; ++nt) {
            Frag bf;
#pragma unroll
            for (int p = 0; p < 8; ++p)
                bf.u[p] = bs_dw[((nt * 16 + hl) << 4) + (kpair2(p, L) >> 1)];
            acc[0][nt] = __builtin_amdgcn_wmma_f32_16x16x32_bf16(
                false, a0.v, false, bf.v, (short)0, acc[0][nt], false, false);
            acc[1][nt] = __builtin_amdgcn_wmma_f32_16x16x32_bf16(
                false, a1.v, false, bf.v, (short)0, acc[1][nt], false, false);
        }
    };

#ifdef USE_TDM
    // Generic LDS pointer: low 32 bits are the LDS byte offset (shared aperture).
    const unsigned lds0 = (unsigned)(unsigned long long)(uintptr_t)&Bs[0][0];
    const unsigned lds1 = (unsigned)(unsigned long long)(uintptr_t)&Bs[1][0];
    const unsigned long long wbase =
        (unsigned long long)(uintptr_t)w + (unsigned long long)(n0 * EE) * 2ull;

    if (wave == 0) tdm_load_tile(lds0, wbase);                 // tile 0 -> buf 0
    for (int i = 0; i < EE / 32; ++i) {
        const int k0 = i * 32;
        if (wave == 0) __builtin_amdgcn_s_wait_tensorcnt(0);   // tile i landed
        __syncthreads();                                       // publish to all waves
        if (wave == 0 && (i + 1) < EE / 32)                    // prefetch tile i+1
            tdm_load_tile((i & 1) ? lds0 : lds1,
                          wbase + (unsigned long long)(k0 + 32) * 2ull);
        compute_step(k0, (const unsigned int*)&Bs[i & 1][0]);
        __syncthreads();                                       // done with buf i&1
    }
#else
    unsigned int* bs_dw_w = (unsigned int*)&Bs[0][0];
    const unsigned int* wdw = (const unsigned int*)w;
    for (int i = 0; i < EE / 32; ++i) {
        const int k0 = i * 32;
#pragma unroll
        for (int q = 0; q < 4; ++q) {
            int dIdx = tid + q * 256;            // 0..1023 dwords of the 64x32 tile
            int row = dIdx >> 4, cdw = dIdx & 15;
            bs_dw_w[(row << 4) + cdw] = wdw[(((n0 + row) * EE + k0) >> 1) + cdw];
        }
        __syncthreads();
        if (k0 + 32 < EE)
            __builtin_prefetch((const void*)(adw0 + ((k0 + 32) >> 1)), 0, 1);
        compute_step(k0, (const unsigned int*)&Bs[0][0]);
        __syncthreads();
    }
#endif

#pragma unroll
    for (int g = 0; g < 2; ++g) {
#pragma unroll
        for (int nt = 0; nt < 4; ++nt) {
            const int n = n0 + nt * 16 + hl;
            const float bn = bias[n];
#pragma unroll
            for (int v = 0; v < 8; ++v) {
                const int m = m0 + g * 16 + v + (L << 3);
                out[m * EE + n] = f2bf((acc[g][nt][v] + bn) * scalef);
            }
        }
    }
}

// ---------- local (sliding-window + prepended global) attention ----------
// grid: (W/16 = 16 query tiles, nc = 16 chunks, B*H = 24); block 256 = 8 waves.
// keys: jk in [0,1024): jk<16 -> global token jk*256; 16<=jk<784 -> window key
// (key_pos = c*256 + (jk-16) - 256); jk>=784 -> padding (masked out).
__global__ void __launch_bounds__(256)
attn_local_kernel(const unsigned short* __restrict__ qb, const unsigned short* __restrict__ kbuf,
                  const unsigned short* __restrict__ vb, float* __restrict__ out) {
    __shared__ unsigned short Pst[8][16 * 32];   // per-wave P tile (A-frag layout)
    __shared__ float accS[8][4][8 * 32];         // per-wave partial outputs
    __shared__ float rmS[8][16];
    __shared__ float rlS[8][16];

    const int tid = threadIdx.x;
    const int wave = tid >> 5, lane = tid & 31, L = lane >> 4, hl = lane & 15;
    const int z = blockIdx.z;
    const int b = z / HH, h = z - b * HH;
    const int c = blockIdx.y;
    const int i0 = blockIdx.x * 16;

    // Q fragments (row = query, K-dim = head dim 64 -> 2 fragments)
    const int qrow = c * WW + i0 + hl;
    const unsigned int* qdw =
        (const unsigned int*)qb + (((b * TT + qrow) * EE + h * DH) >> 1);
    Frag a0, a1;
#pragma unroll
    for (int p = 0; p < 8; ++p) {
        a0.u[p] = qdw[kpair2(p, L) >> 1];
        a1.u[p] = qdw[(32 + kpair2(p, L)) >> 1];
    }

    v8f acc0 = {}, acc1 = {}, acc2 = {}, acc3 = {};
    float rm[8], rl[8];
#pragma unroll
    for (int v = 0; v < 8; ++v) { rm[v] = -1e30f; rl[v] = 0.f; }

    const int kb0 = wave * 128;
    for (int step = 0; step < 4; ++step) {
        const int kbase = kb0 + step * 32;
        float sv[2][8];
#pragma unroll
        for (int sub = 0; sub < 2; ++sub) {
            const int jk = kbase + sub * 16 + hl;
            int t = (jk < 16) ? (jk << 8) : (c * WW + (jk - 16) - WW);
            const int tc = (t < 0) ? 0 : ((t > TT - 1) ? TT - 1 : t);
            const unsigned int* kdw =
                (const unsigned int*)kbuf + (((b * TT + tc) * EE + h * DH) >> 1);
            Frag bk0, bk1;
#pragma unroll
            for (int p = 0; p < 8; ++p) {
                bk0.u[p] = kdw[kpair2(p, L) >> 1];
                bk1.u[p] = kdw[(32 + kpair2(p, L)) >> 1];
            }
            v8f s = {};
            s = __builtin_amdgcn_wmma_f32_16x16x32_bf16(false, a0.v, false, bk0.v,
                                                        (short)0, s, false, false);
            s = __builtin_amdgcn_wmma_f32_16x16x32_bf16(false, a1.v, false, bk1.v,
                                                        (short)0, s, false, false);
            const bool isglob = (jk < 16);
            const bool inseq = (t >= 0) && (t < TT);
            const bool isg_tok = ((t & 255) == 0);
#pragma unroll
            for (int v = 0; v < 8; ++v) {
                float xsc = s[v];
                if (!isglob) {
                    const int j = jk - 16;
                    const int i = i0 + v + (L << 3);
                    const bool valid = inseq && (jk < 784) && (j >= i) && (j <= i + 2 * WW);
                    xsc = valid ? (isg_tok ? xsc - 10000.f : xsc) : -1e9f;
                }
                sv[sub][v] = xsc;
            }
        }
        // online-softmax update (per output row = v + 8*L)
#pragma unroll
        for (int v = 0; v < 8; ++v) {
            float nm = fmaxf(rm[v], fmaxf(rmax16(sv[0][v]), rmax16(sv[1][v])));
            float sc = __expf(rm[v] - nm);
            rm[v] = nm;
            rl[v] *= sc;
            acc0[v] *= sc; acc1[v] *= sc; acc2[v] *= sc; acc3[v] *= sc;
            float p0 = __expf(sv[0][v] - nm);
            float p1 = __expf(sv[1][v] - nm);
            rl[v] += rsum16(p0) + rsum16(p1);
            const int r = v + (L << 3);
            Pst[wave][(r << 5) + hl] = f2bf(p0);
            Pst[wave][(r << 5) + 16 + hl] = f2bf(p1);
        }
        // P fragment (A layout) from this wave's private LDS strip
        Frag pf;
        const unsigned int* pdw = (const unsigned int*)&Pst[wave][0];
#pragma unroll
        for (int p = 0; p < 8; ++p)
            pf.u[p] = pdw[((hl << 5) + kpair2(p, L)) >> 1];
        // V fragments (B layout: lane = d column, elements = keys)
#pragma unroll
        for (int nt = 0; nt < 4; ++nt) {
            const int d = nt * 16 + hl;
            Frag vf;
#pragma unroll
            for (int e = 0; e < 16; ++e) {
                const int jk = kbase + kloc(e, L);
                int t = (jk < 16) ? (jk << 8) : (c * WW + (jk - 16) - WW);
                t = (t < 0) ? 0 : ((t > TT - 1) ? TT - 1 : t);
                vf.us[e] = vb[(b * TT + t) * EE + h * DH + d];
            }
            v8f* ap = (nt == 0) ? &acc0 : (nt == 1) ? &acc1 : (nt == 2) ? &acc2 : &acc3;
            *ap = __builtin_amdgcn_wmma_f32_16x16x32_bf16(false, pf.v, false, vf.v,
                                                          (short)0, *ap, false, false);
        }
    }

    // export per-wave partial state
#pragma unroll
    for (int v = 0; v < 8; ++v) {
        accS[wave][0][(v << 5) + lane] = acc0[v];
        accS[wave][1][(v << 5) + lane] = acc1[v];
        accS[wave][2][(v << 5) + lane] = acc2[v];
        accS[wave][3][(v << 5) + lane] = acc3[v];
    }
    if (hl == 0) {
#pragma unroll
        for (int v = 0; v < 8; ++v) {
            rmS[wave][v + (L << 3)] = rm[v];
            rlS[wave][v + (L << 3)] = rl[v];
        }
    }
    __syncthreads();

    // merge 8 wave partials (split-K flash merge), normalize, store f32
    for (int o = tid; o < 16 * DH; o += 256) {
        const int r = o >> 6, d = o & 63;
        const int nt = d >> 4, v = r & 7, ln = (d & 15) + ((r >> 3) << 4);
        float M = -1e30f;
#pragma unroll
        for (int w = 0; w < 8; ++w) M = fmaxf(M, rmS[w][r]);
        float Ls = 0.f, val = 0.f;
#pragma unroll
        for (int w = 0; w < 8; ++w) {
            const float e = __expf(rmS[w][r] - M);
            Ls += rlS[w][r] * e;
            val += accS[w][nt][(v << 5) + ln] * e;
        }
        out[(b * TT + (c * WW + i0 + r)) * EE + h * DH + d] = val / Ls;
    }
}

// ---------- global-token attention (overwrites rows at GLOBAL_IDX) ----------
// grid: B*H blocks; one block handles the 16 global queries vs all 4096 keys.
__global__ void __launch_bounds__(256)
attn_global_kernel(const unsigned short* __restrict__ qgb, const unsigned short* __restrict__ kgb,
                   const unsigned short* __restrict__ vgb, float* __restrict__ out) {
    __shared__ unsigned short Pst[8][16 * 32];
    __shared__ float accS[8][4][8 * 32];
    __shared__ float rmS[8][16];
    __shared__ float rlS[8][16];

    const int tid = threadIdx.x;
    const int wave = tid >> 5, lane = tid & 31, L = lane >> 4, hl = lane & 15;
    const int z = blockIdx.x;
    const int b = z / HH, h = z - b * HH;

    const int qrow = hl << 8;    // global token hl -> t = hl*256
    const unsigned int* qdw =
        (const unsigned int*)qgb + (((b * TT + qrow) * EE + h * DH) >> 1);
    Frag a0, a1;
#pragma unroll
    for (int p = 0; p < 8; ++p) {
        a0.u[p] = qdw[kpair2(p, L) >> 1];
        a1.u[p] = qdw[(32 + kpair2(p, L)) >> 1];
    }

    v8f acc0 = {}, acc1 = {}, acc2 = {}, acc3 = {};
    float rm[8], rl[8];
#pragma unroll
    for (int v = 0; v < 8; ++v) { rm[v] = -1e30f; rl[v] = 0.f; }

    const int kb0 = wave * 512;
    for (int step = 0; step < 16; ++step) {
        const int kbase = kb0 + step * 32;
        float sv[2][8];
#pragma unroll
        for (int sub = 0; sub < 2; ++sub) {
            const int t = kbase + sub * 16 + hl;
            const unsigned int* kdw =
                (const unsigned int*)kgb + (((b * TT + t) * EE + h * DH) >> 1);
            Frag bk0, bk1;
#pragma unroll
            for (int p = 0; p < 8; ++p) {
                bk0.u[p] = kdw[kpair2(p, L) >> 1];
                bk1.u[p] = kdw[(32 + kpair2(p, L)) >> 1];
            }
            v8f s = {};
            s = __builtin_amdgcn_wmma_f32_16x16x32_bf16(false, a0.v, false, bk0.v,
                                                        (short)0, s, false, false);
            s = __builtin_amdgcn_wmma_f32_16x16x32_bf16(false, a1.v, false, bk1.v,
                                                        (short)0, s, false, false);
#pragma unroll
            for (int v = 0; v < 8; ++v) sv[sub][v] = s[v];
        }
#pragma unroll
        for (int v = 0; v < 8; ++v) {
            float nm = fmaxf(rm[v], fmaxf(rmax16(sv[0][v]), rmax16(sv[1][v])));
            float sc = __expf(rm[v] - nm);
            rm[v] = nm;
            rl[v] *= sc;
            acc0[v] *= sc; acc1[v] *= sc; acc2[v] *= sc; acc3[v] *= sc;
            float p0 = __expf(sv[0][v] - nm);
            float p1 = __expf(sv[1][v] - nm);
            rl[v] += rsum16(p0) + rsum16(p1);
            const int r = v + (L << 3);
            Pst[wave][(r << 5) + hl] = f2bf(p0);
            Pst[wave][(r << 5) + 16 + hl] = f2bf(p1);
        }
        Frag pf;
        const unsigned int* pdw = (const unsigned int*)&Pst[wave][0];
#pragma unroll
        for (int p = 0; p < 8; ++p)
            pf.u[p] = pdw[((hl << 5) + kpair2(p, L)) >> 1];
#pragma unroll
        for (int nt = 0; nt < 4; ++nt) {
            const int d = nt * 16 + hl;
            Frag vf;
#pragma unroll
            for (int e = 0; e < 16; ++e) {
                const int t = kbase + kloc(e, L);
                vf.us[e] = vgb[(b * TT + t) * EE + h * DH + d];
            }
            v8f* ap = (nt == 0) ? &acc0 : (nt == 1) ? &acc1 : (nt == 2) ? &acc2 : &acc3;
            *ap = __builtin_amdgcn_wmma_f32_16x16x32_bf16(false, pf.v, false, vf.v,
                                                          (short)0, *ap, false, false);
        }
    }

#pragma unroll
    for (int v = 0; v < 8; ++v) {
        accS[wave][0][(v << 5) + lane] = acc0[v];
        accS[wave][1][(v << 5) + lane] = acc1[v];
        accS[wave][2][(v << 5) + lane] = acc2[v];
        accS[wave][3][(v << 5) + lane] = acc3[v];
    }
    if (hl == 0) {
#pragma unroll
        for (int v = 0; v < 8; ++v) {
            rmS[wave][v + (L << 3)] = rm[v];
            rlS[wave][v + (L << 3)] = rl[v];
        }
    }
    __syncthreads();

    for (int o = tid; o < 16 * DH; o += 256) {
        const int r = o >> 6, d = o & 63;
        const int nt = d >> 4, v = r & 7, ln = (d & 15) + ((r >> 3) << 4);
        float M = -1e30f;
#pragma unroll
        for (int w = 0; w < 8; ++w) M = fmaxf(M, rmS[w][r]);
        float Ls = 0.f, val = 0.f;
#pragma unroll
        for (int w = 0; w < 8; ++w) {
            const float e = __expf(rmS[w][r] - M);
            Ls += rlS[w][r] * e;
            val += accS[w][nt][(v << 5) + ln] * e;
        }
        out[(b * TT + (r << 8)) * EE + h * DH + d] = val / Ls;
    }
}

// ---------- host launcher ----------
extern "C" void kernel_launch(void* const* d_in, const int* in_sizes, int n_in,
                              void* d_out, int out_size, void* d_ws, size_t ws_size,
                              hipStream_t stream) {
    (void)in_sizes; (void)n_in; (void)out_size; (void)ws_size;

    const float* x   = (const float*)d_in[0];
    const float* Wq  = (const float*)d_in[1];  const float* bq  = (const float*)d_in[2];
    const float* Wk  = (const float*)d_in[3];  const float* bk  = (const float*)d_in[4];
    const float* Wv  = (const float*)d_in[5];  const float* bv  = (const float*)d_in[6];
    const float* Wqg = (const float*)d_in[7];  const float* bqg = (const float*)d_in[8];
    const float* Wkg = (const float*)d_in[9];  const float* bkg = (const float*)d_in[10];
    const float* Wvg = (const float*)d_in[11]; const float* bvg = (const float*)d_in[12];
    float* out = (float*)d_out;

    const size_t MTOT = (size_t)BB * TT;           // 8192
    unsigned short* ws   = (unsigned short*)d_ws;
    unsigned short* xb   = ws;                                     // 8192*768
    unsigned short* wt   = xb + MTOT * EE;                         // 6*768*768
    unsigned short* proj = wt + (size_t)6 * EE * EE;               // 6*8192*768

    const int nx = (int)(MTOT * EE);
    conv_x_kernel<<<(nx + 255) / 256, 256, 0, stream>>>(x, xb, nx);
    conv_w_kernel<<<dim3((EE * EE) / 256, 6), 256, 0, stream>>>(Wq, Wk, Wv, Wqg, Wkg, Wvg, wt);

    proj_gemm_kernel<<<dim3(EE / 64, (int)(MTOT / 256), 6), 256, 0, stream>>>(
        xb, wt, proj, bq, bk, bv, bqg, bkg, bvg);

    const unsigned short* qb  = proj;
    const unsigned short* kbp = proj + 1 * MTOT * EE;
    const unsigned short* vbp = proj + 2 * MTOT * EE;
    const unsigned short* qgb = proj + 3 * MTOT * EE;
    const unsigned short* kgb = proj + 4 * MTOT * EE;
    const unsigned short* vgb = proj + 5 * MTOT * EE;

    attn_local_kernel<<<dim3(WW / 16, NC, BB * HH), 256, 0, stream>>>(qb, kbp, vbp, out);
    attn_global_kernel<<<BB * HH, 256, 0, stream>>>(qgb, kgb, vgb, out);
}
